// MultiHeadAttention_21234318311571
// MI455X (gfx1250) — matrix operand order repaired and checked by reading, not compiled, not run on hardware
//
#include <hip/hip_runtime.h>
#include <hip/hip_bf16.h>

#define B_ 2
#define T_ 2048
#define D_ 1024
#define H_ 16
#define DH_ 64
#define BT_ (B_*T_)

typedef __attribute__((ext_vector_type(16))) _Float16 v16h;
typedef __attribute__((ext_vector_type(8)))  _Float16 v8h;
typedef __attribute__((ext_vector_type(8)))  float    v8f;
typedef __attribute__((ext_vector_type(4)))  unsigned int u32x4;
typedef __attribute__((ext_vector_type(8)))  int i32x8;
typedef __attribute__((ext_vector_type(4)))  int i32x4;

#if __has_builtin(__builtin_amdgcn_tensor_load_to_lds) && __has_builtin(__builtin_amdgcn_s_wait_tensorcnt)
#define USE_TDM 1
#else
#define USE_TDM 0
#endif

static __device__ inline v16h cat8(v8h lo, v8h hi) {
  return __builtin_shufflevector(lo, hi, 0,1,2,3,4,5,6,7,8,9,10,11,12,13,14,15);
}
// 16 contiguous halves (32B) -> B fragment slice for one lane
static __device__ inline v16h ld16h(const _Float16* p) {
  const v8h* q = (const v8h*)p;
  return cat8(q[0], q[1]);
}
static __device__ inline v8f wmma_f16(v16h a, v16h b, v8f c) {
  // (neg_a, A, neg_b, B, c_mod, C, reuse_a, reuse_b)
  return __builtin_amdgcn_wmma_f32_16x16x32_f16(false, a, false, b, (short)0, c, false, false);
}
static __device__ inline float halfmax(float v) {
  v = fmaxf(v, __shfl_xor(v, 1, 32));
  v = fmaxf(v, __shfl_xor(v, 2, 32));
  v = fmaxf(v, __shfl_xor(v, 4, 32));
  v = fmaxf(v, __shfl_xor(v, 8, 32));
  return v;
}
static __device__ inline float halfsum(float v) {
  v += __shfl_xor(v, 1, 32);
  v += __shfl_xor(v, 2, 32);
  v += __shfl_xor(v, 4, 32);
  v += __shfl_xor(v, 8, 32);
  return v;
}

#if USE_TDM
// TDM 2D tile load: 16 rows x 32 f16 from a [1024 x 1024] f16 tensor (row
// stride 1024 elems) into LDS with 4-DWORD pad every 16 DWORDs (80B rows).
static __device__ inline void tdm_issue(unsigned lds_off, unsigned long long gaddr) {
  u32x4 g0;
  g0[0] = 1u;                                                 // count=1, user
  g0[1] = lds_off;                                            // LDS byte addr
  g0[2] = (unsigned)gaddr;                                    // global addr lo
  g0[3] = (unsigned)((gaddr >> 32) & 0x01FFFFFFu) | (2u << 30); // addr hi | type=2
  i32x8 g1;
  g1[0] = (1 << 16) | (1 << 20) | (3 << 22) | (3 << 25); // 2B elems, pad: 4dw/16dw
  g1[1] = (int)(1024u << 16);   // tensor_dim0 = 1024 (low16 at bits 63:48)
  g1[2] = (int)(1024u << 16);   // dim0 hi=0 | tensor_dim1 = 1024 (low16)
  g1[3] = (int)(32u << 16);     // dim1 hi=0 | tile_dim0 = 32
  g1[4] = 16;                   // tile_dim1 = 16, tile_dim2 = 0
  g1[5] = 1024;                 // tensor_dim0_stride = 1024 elems
  g1[6] = 0;
  g1[7] = 0;
  i32x4 gz4 = {0, 0, 0, 0};
  i32x8 gz8 = {0, 0, 0, 0, 0, 0, 0, 0};
  __builtin_amdgcn_tensor_load_to_lds(g0, g1, gz4, gz4, gz8, 0);
}
#endif

// ---------------- LayerNorm (f32 in -> f16 out), one wave per row ----------
__global__ void ln_kernel(const float* __restrict__ x, const float* __restrict__ gamma,
                          const float* __restrict__ beta, _Float16* __restrict__ y) {
  int wid  = (blockIdx.x * blockDim.x + threadIdx.x) >> 5;
  int lane = threadIdx.x & 31;
  if (wid >= BT_) return;
  const float* row = x + (size_t)wid * D_;
  float s = 0.f, s2 = 0.f;
  for (int i = lane; i < D_; i += 32) { float v = row[i]; s += v; s2 += v * v; }
  for (int m = 16; m >= 1; m >>= 1) { s += __shfl_xor(s, m, 32); s2 += __shfl_xor(s2, m, 32); }
  float mean = s * (1.f / D_);
  float var  = s2 * (1.f / D_) - mean * mean;
  float rstd = rsqrtf(var + 1e-5f);
  _Float16* o = y + (size_t)wid * D_;
  for (int i = lane; i < D_; i += 32)
    o[i] = (_Float16)((row[i] - mean) * rstd * gamma[i] + beta[i]);
}

// ------------- weight transpose+convert: Wt[n][k] = (f16) W[k][n] ----------
__global__ void wt_kernel(const float* __restrict__ W, _Float16* __restrict__ Wt) {
  int idx = blockIdx.x * blockDim.x + threadIdx.x;   // = n*D + k
  int n = idx >> 10, k = idx & (D_ - 1);
  Wt[idx] = (_Float16)W[(size_t)k * D_ + n];
}

// ------ WMMA GEMM: [4096 x 1024] f16 x Wt (f16, N-major), 64x64 block tile -
// 4 waves; wave w computes rows [bm*64+16w, +16) x all 64 cols (4 C frags).
// B tile (64 cols x 32 k) staged in LDS via TDM (80B padded rows), dbl-buffered.
// mode 0: Q -> f16 [B,H,T,DH]   mode 1: K -> f16 [B,H,T,DH]
// mode 2: V -> f16 [B,H,DH,T]   mode 3: f32 out [BT,D]
#define ROWB 80           // LDS bytes per B-tile row (64 data + 16 pad)
#define BUFB (64 * ROWB)  // 5120 bytes per buffer
__global__ void __launch_bounds__(128) gemm_kernel(
    const _Float16* __restrict__ A, const _Float16* __restrict__ Wt,
    const float* __restrict__ bias, void* __restrict__ out, int mode) {
  __shared__ __align__(16) char bsm[2 * BUFB];
  int lane = threadIdx.x & 31;
  int w    = threadIdx.x >> 5;
  int l16  = lane & 15, g = lane >> 4;
  int bm = blockIdx.x / (D_ / 64);          // 0..63
  int bn = blockIdx.x % (D_ / 64);          // 0..15
  int m0 = bm * 64 + 16 * w;
  int n0 = bn * 64;

  v8f c[4] = {{}, {}, {}, {}};
  const _Float16* arow = A + (size_t)(m0 + l16) * D_;

#if USE_TDM
  unsigned lds_q = ((unsigned)(uintptr_t)(&bsm[0])) + (unsigned)(w * 16 * ROWB);
  unsigned long long gbase =
      (unsigned long long)(uintptr_t)(Wt + (size_t)(n0 + 16 * w) * D_);
  tdm_issue(lds_q, gbase);                          // k-step 0 -> buf 0
  // A fragment prefetch for k-step 0
  v16h a = cat8(((const v8h*)(arow + 16 * g))[0],
                ((const v8h*)(arow + 8 + 16 * g))[0]);
  #pragma unroll 1
  for (int i = 0; i < D_ / 32; i++) {
    __builtin_amdgcn_s_wait_tensorcnt(0);           // my quarter of buf (i&1) done
    __syncthreads();                                // whole tile ready; prev compute done
    if (i + 1 < D_ / 32)
      tdm_issue(lds_q + (unsigned)(((i + 1) & 1) * BUFB),
                gbase + (unsigned long long)(i + 1) * 64u);
    const char* bufc = &bsm[(i & 1) * BUFB];
    v16h a_cur = a;
    if (i + 1 < D_ / 32) {                          // prefetch next A fragment
      int kk = (i + 1) * 32;
      a = cat8(((const v8h*)(arow + kk + 16 * g))[0],
               ((const v8h*)(arow + kk + 8 + 16 * g))[0]);
    }
    #pragma unroll
    for (int cc = 0; cc < 4; cc++) {
      const _Float16* bp = (const _Float16*)(bufc + (16 * cc + l16) * ROWB + 32 * g);
      c[cc] = wmma_f16(a_cur, ld16h(bp), c[cc]);
    }
  }
#else
  // fallback: cooperative copy, single buffer
  #pragma unroll 1
  for (int i = 0; i < D_ / 32; i++) {
    __syncthreads();
    {
      int r = threadIdx.x >> 1, hh = threadIdx.x & 1;
      const v8h* src = (const v8h*)(Wt + (size_t)(n0 + r) * D_ + i * 32 + 16 * hh);
      v8h* dst = (v8h*)(&bsm[r * ROWB + hh * 32]);
      dst[0] = src[0];
      dst[1] = src[1];
    }
    __syncthreads();
    int kk = i * 32;
    v16h a = cat8(((const v8h*)(arow + kk + 16 * g))[0],
                  ((const v8h*)(arow + kk + 8 + 16 * g))[0]);
    #pragma unroll
    for (int cc = 0; cc < 4; cc++) {
      const _Float16* bp = (const _Float16*)(&bsm[(16 * cc + l16) * ROWB + 32 * g]);
      c[cc] = wmma_f16(a, ld16h(bp), c[cc]);
    }
  }
#endif

  if (mode == 3) {
    float* O = (float*)out;
    #pragma unroll
    for (int cc = 0; cc < 4; cc++) {
      float bval = bias[n0 + 16 * cc + l16];
      #pragma unroll
      for (int j = 0; j < 8; j++)
        O[(size_t)(m0 + j + 8 * g) * D_ + n0 + 16 * cc + l16] = c[cc][j] + bval;
    }
  } else {
    _Float16* O = (_Float16*)out;
    #pragma unroll
    for (int cc = 0; cc < 4; cc++) {
      int col = n0 + 16 * cc + l16;
      int h = col >> 6, dh = col & 63;
      float bval = bias[col];
      #pragma unroll
      for (int j = 0; j < 8; j++) {
        int row = m0 + j + 8 * g;              // 0..4095
        int bb  = row >> 11, tt = row & (T_ - 1);
        float v = c[cc][j] + bval;
        size_t off;
        if (mode == 2) off = ((size_t)(bb * H_ + h) * DH_ + dh) * T_ + tt;   // V^T
        else           off = ((size_t)(bb * H_ + h) * T_ + tt) * DH_ + dh;   // Q / K
        O[off] = (_Float16)v;
      }
    }
  }
}

// -------- fused attention: per wave one (b,h, 16-query tile); 2-pass -------
__global__ void attn_kernel(const _Float16* __restrict__ Qf, const _Float16* __restrict__ Kf,
                            const _Float16* __restrict__ Vt, const int* __restrict__ lens,
                            float* __restrict__ viz, _Float16* __restrict__ AO) {
  __shared__ __align__(16) _Float16 pbuf[4][16 * 32]; // per-wave P staging (C->A transpose)
  int wslot = threadIdx.x >> 5;
  int lane  = threadIdx.x & 31;
  int wid   = blockIdx.x * (blockDim.x >> 5) + wslot;
  const int QT = T_ / 16;                 // 128 query tiles
  int qt = wid & (QT - 1);
  int bh = wid / QT;                      // 0..31
  if (bh >= B_ * H_) return;
  int b = bh / H_, h = bh % H_;
  int len = lens[b];
  int q0 = qt * 16;
  int l16 = lane & 15, g = lane >> 4;
  const float scale = 0.125f;             // 1/sqrt(64)

  const _Float16* Qb = Qf + (size_t)bh * T_ * DH_;
  const _Float16* Kb = Kf + (size_t)bh * T_ * DH_;
  const _Float16* Vb = Vt + (size_t)bh * DH_ * T_;

  // Q A-fragments: frag ka covers dh in [32ka, 32ka+32)
  v16h qa[2];
  {
    const _Float16* qrow = Qb + (size_t)(q0 + l16) * DH_;
    #pragma unroll
    for (int ka = 0; ka < 2; ka++) {
      const v8h* p0 = (const v8h*)(qrow + 32 * ka + 16 * g);
      const v8h* p1 = (const v8h*)(qrow + 32 * ka + 8 + 16 * g);
      qa[ka] = cat8(p0[0], p1[0]);
    }
  }

  int kmax   = min(q0 + 15, len - 1);     // last key any row in tile can see
  int ktiles = (kmax >> 4) + 1;           // key tiles with any valid entry

  float rmax[8], rsum[8];
  #pragma unroll
  for (int j = 0; j < 8; j++) { rmax[j] = -1e30f; rsum[j] = 0.f; }

  // ---- pass 1: row max / sum of exp (online) ----
  #pragma unroll 1
  for (int kt = 0; kt < ktiles; kt++) {
    int k0 = kt * 16;
    v8f s = {};
    const _Float16* krow = Kb + (size_t)(k0 + l16) * DH_;
    #pragma unroll
    for (int kb = 0; kb < 2; kb++)
      s = wmma_f16(qa[kb], ld16h(krow + 32 * kb + 16 * g), s);
    #pragma unroll
    for (int j = 0; j < 8; j++) {
      int q   = q0 + j + 8 * g;
      int key = k0 + l16;
      float sv = (key <= q && key < len) ? s[j] * scale : -3.0e38f;
      float mx = halfmax(sv);
      float nm = fmaxf(rmax[j], mx);
      float e  = (sv > -1e37f) ? __expf(sv - nm) : 0.f;
      float es = halfsum(e);
      rsum[j] = rsum[j] * __expf(rmax[j] - nm) + es;
      rmax[j] = nm;
    }
  }
  float rinv[8];
  #pragma unroll
  for (int j = 0; j < 8; j++) rinv[j] = 1.f / rsum[j];

  // ---- pass 2: write normalized P to viz, accumulate O += P @ V ----
  v8f oc[4] = {{}, {}, {}, {}};
  _Float16* pb = pbuf[wslot];
  float* vizb = viz + (size_t)bh * T_ * T_;
  #pragma unroll 1
  for (int kg = 0; kg < T_ / 32; kg++) {            // 32-key groups
    bool any = (kg * 32) < (ktiles * 16);
    #pragma unroll
    for (int sub = 0; sub < 2; sub++) {
      int kt = kg * 2 + sub;
      int k0 = kt * 16;
      float pv[8];
      if (kt < ktiles) {                            // wave-uniform branch
        v8f s = {};
        const _Float16* krow = Kb + (size_t)(k0 + l16) * DH_;
        #pragma unroll
        for (int kb = 0; kb < 2; kb++)
          s = wmma_f16(qa[kb], ld16h(krow + 32 * kb + 16 * g), s);
        #pragma unroll
        for (int j = 0; j < 8; j++) {
          int q   = q0 + j + 8 * g;
          int key = k0 + l16;
          pv[j] = (key <= q && key < len) ? __expf(s[j] * scale - rmax[j]) * rinv[j] : 0.f;
        }
      } else {
        #pragma unroll
        for (int j = 0; j < 8; j++) pv[j] = 0.f;
      }
      #pragma unroll
      for (int j = 0; j < 8; j++) {
        int m = j + 8 * g;
        vizb[(size_t)(q0 + m) * T_ + k0 + l16] = pv[j];
        pb[m * 32 + sub * 16 + l16] = (_Float16)pv[j];
      }
    }
    if (any) {                                      // wave-uniform
      // P A-fragment from LDS (same-wave DS ops are in-order; no barrier needed)
      const v8h* pl0 = (const v8h*)(pb + l16 * 32 + 16 * g);
      const v8h* pl1 = (const v8h*)(pb + l16 * 32 + 8 + 16 * g);
      v16h pa = cat8(pl0[0], pl1[0]);
      #pragma unroll
      for (int c2 = 0; c2 < 4; c2++) {
        const _Float16* vcol = Vb + (size_t)(16 * c2 + l16) * T_ + kg * 32 + 16 * g;
        oc[c2] = wmma_f16(pa, ld16h(vcol), oc[c2]);
      }
    }
  }

  // store O tile to AO [BT, D] f16 (row = b*T+t, col = h*64 + dh)
  #pragma unroll
  for (int c2 = 0; c2 < 4; c2++) {
    #pragma unroll
    for (int j = 0; j < 8; j++) {
      int m = j + 8 * g;
      size_t row = (size_t)b * T_ + q0 + m;
      AO[row * D_ + h * DH_ + 16 * c2 + l16] = (_Float16)oc[c2][j];
    }
  }
}

extern "C" void kernel_launch(void* const* d_in, const int* in_sizes, int n_in,
                              void* d_out, int out_size, void* d_ws, size_t ws_size,
                              hipStream_t stream) {
  (void)in_sizes; (void)n_in; (void)out_size; (void)ws_size;
  const float* xq    = (const float*)d_in[0];
  const float* xk    = (const float*)d_in[1];
  const float* xv    = (const float*)d_in[2];
  const int*   lens  = (const int*)d_in[3];
  const float* Wq    = (const float*)d_in[4];
  const float* bq    = (const float*)d_in[5];
  const float* Wk    = (const float*)d_in[6];
  const float* bk    = (const float*)d_in[7];
  const float* Wv    = (const float*)d_in[8];
  const float* bv    = (const float*)d_in[9];
  const float* Wo    = (const float*)d_in[10];
  const float* bo    = (const float*)d_in[11];
  const float* gamma = (const float*)d_in[12];
  const float* beta  = (const float*)d_in[13];

  float* out = (float*)d_out;
  float* viz = out + (size_t)BT_ * D_;

  // workspace carve-up (all f16): 64 MB total
  _Float16* p   = (_Float16*)d_ws;
  _Float16* lnq = p; p += (size_t)BT_ * D_;
  _Float16* lnk = p; p += (size_t)BT_ * D_;
  _Float16* lnv = p; p += (size_t)BT_ * D_;
  _Float16* WtQ = p; p += (size_t)D_ * D_;
  _Float16* WtK = p; p += (size_t)D_ * D_;
  _Float16* WtV = p; p += (size_t)D_ * D_;
  _Float16* WtO = p; p += (size_t)D_ * D_;
  _Float16* Qf  = p; p += (size_t)B_ * H_ * T_ * DH_;
  _Float16* Kf  = p; p += (size_t)B_ * H_ * T_ * DH_;
  _Float16* Vt  = p; p += (size_t)B_ * H_ * T_ * DH_;
  _Float16* AO  = p; p += (size_t)BT_ * D_;

  dim3 blk(128);  // 4 waves per block (wave32)

  ln_kernel<<<BT_ / 4, blk, 0, stream>>>(xq, gamma, beta, lnq);
  ln_kernel<<<BT_ / 4, blk, 0, stream>>>(xk, gamma, beta, lnk);
  ln_kernel<<<BT_ / 4, blk, 0, stream>>>(xv, gamma, beta, lnv);

  wt_kernel<<<D_ * D_ / 256, 256, 0, stream>>>(Wq, WtQ);
  wt_kernel<<<D_ * D_ / 256, 256, 0, stream>>>(Wk, WtK);
  wt_kernel<<<D_ * D_ / 256, 256, 0, stream>>>(Wv, WtV);
  wt_kernel<<<D_ * D_ / 256, 256, 0, stream>>>(Wo, WtO);

  // 64 x 16 block tiles of 64x64 -> 1024 blocks of 4 waves
  gemm_kernel<<<1024, blk, 0, stream>>>(lnq, WtQ, bq, Qf, 0);
  gemm_kernel<<<1024, blk, 0, stream>>>(lnk, WtK, bk, Kf, 1);
  gemm_kernel<<<1024, blk, 0, stream>>>(lnv, WtV, bv, Vt, 2);

  // B*H*(T/16) = 4096 waves -> 1024 blocks
  attn_kernel<<<1024, blk, 0, stream>>>(Qf, Kf, Vt, lens, viz, AO);

  gemm_kernel<<<1024, blk, 0, stream>>>(AO, WtO, bo, out, 3);
}
